// Network_57509612094122
// MI455X (gfx1250) — matrix operand order repaired and checked by reading, not compiled, not run on hardware
//
#include <hip/hip_runtime.h>

// Fused PointConv-style block for MI455X (gfx1250, wave32, WMMA).
// One kernel; per-wave 16-point tiles processed as two 8-point groups
// (halves LDS -> doubles resident waves per WGP for latency hiding).
// All channel-mixing via v_wmma_f32_16x16x32_f16 with f32 accumulation.

typedef __attribute__((ext_vector_type(16))) _Float16 v16h;
typedef __attribute__((ext_vector_type(8)))  float    v8f;

#define WAVES 4
#define TPB   (WAVES * 32)
#define NPTS  65536
#define BATCH 2
#define KNB   16
#define SA 36    // tA row stride (halfs), holds up to [32][36]
#define SW 18    // tW row stride
#define SN 516   // tN row stride (512 + pad)
#define SL 33    // staged lin_w col stride

struct P {
  const float *feature, *xyz;
  const float *mlp1_w, *mlp1_b, *mlp1_bn;
  const float *wn0_w, *wn0_b, *wn0_bn;
  const float *wn1_w, *wn1_b, *wn1_bn;
  const float *wn2_w, *wn2_b, *wn2_bn;
  const float *fc_w;
  const float *c0_w, *c0_b, *c0_bn;
  const float *c1_w, *c1_b, *c1_bn;
  const float *c2_w, *c2_b, *c2_bn;
  const float *c3_w, *c3_b, *c3_bn;
  const float *lin_w, *lin_b, *lin_bn;
  const float *mlp2_w, *mlp2_b, *mlp2_bn;
  const float *sc_w, *sc_b, *sc_bn;
  const int   *nidx;
  float       *out;
};

// ---- WMMA wrapper (confirmed 8-arg signature) ----
__device__ __forceinline__ v8f wmma(v16h a, v16h b, v8f c) {
  return __builtin_amdgcn_wmma_f32_16x16x32_f16(false, a, false, b, (short)0, c,
                                                false, false);
}

__device__ __forceinline__ v8f zeroC() {
  v8f z;
#pragma unroll
  for (int i = 0; i < 8; ++i) z[i] = 0.f;
  return z;
}

// A k order used here (contiguous, not AMD's table order): lanes 0-15 hold M=lane; element h -> K h,
// lanes 16-31 hold K 16..31 (matches B k order).
__device__ __forceinline__ int a_k_of(int lane, int h) {
  return ((lane & 16) ? 16 : 0) + h;
}

// A fragment from LDS tile t[m][k] (row stride S); zero beyond cin.
__device__ __forceinline__ v16h load_A(const _Float16 *t, int S, int lane,
                                       int cin) {
  int m = lane & 15;
  v16h a;
#pragma unroll
  for (int h = 0; h < 16; ++h) {
    int k = a_k_of(lane, h);
    a[h] = (k < cin) ? t[m * S + k] : (_Float16)0.0f;
  }
  return a;
}

// B fragment straight from global weight W[cout][cin]: B[k][n] = W[n][k].
// B k order used here (contiguous, as A): lane col = lane&15; lanes 0-15 hold K 0..15, 16-31 hold 16..31.
__device__ __forceinline__ v16h make_B(const float *W, int cin, int cout,
                                       int lane, int colBase) {
  int n = (lane & 15) + colBase;
  v16h b;
#pragma unroll
  for (int h = 0; h < 16; ++h) {
    int k = ((lane & 16) ? 16 : 0) + h;
    float v = (k < cin && n < cout) ? W[n * cin + k] : 0.f;
    b[h] = (_Float16)v;
  }
  return b;
}

// B fragment from LDS tile t[k][n] (row stride S); zero beyond kin.
__device__ __forceinline__ v16h load_Bt(const _Float16 *t, int S, int lane,
                                        int kin) {
  int n = lane & 15;
  v16h b;
#pragma unroll
  for (int h = 0; h < 16; ++h) {
    int k = ((lane & 16) ? 16 : 0) + h;
    b[h] = (k < kin) ? t[k * S + n] : (_Float16)0.0f;
  }
  return b;
}

// f32 C 16x16 layout: lane = col n (dup over halves), VGPR r = row m (+8 hi).
__device__ __forceinline__ void store_C(_Float16 *t, int S, int lane, v8f c) {
  int m0 = (lane & 16) ? 8 : 0;
  int n = lane & 15;
#pragma unroll
  for (int r = 0; r < 8; ++r) t[(m0 + r) * S + n] = (_Float16)c[r];
}

// Fold conv-bias into eval-mode BN: y = s*x + t.
__device__ __forceinline__ void bn_aff(const float *bnp, int C,
                                       const float *bias, int ch, bool valid,
                                       float &s, float &t) {
  if (!valid) { s = 0.f; t = 0.f; return; }
  float g = bnp[ch], be = bnp[C + ch], mu = bnp[2 * C + ch],
        va = bnp[3 * C + ch];
  s = g * rsqrtf(va + 1e-5f);
  float bi = bias ? bias[ch] : 0.f;
  t = s * (bi - mu) + be;
}

__device__ __forceinline__ v8f aff(v8f c, float s, float t, bool relu) {
  v8f r;
#pragma unroll
  for (int i = 0; i < 8; ++i) {
    float x = c[i] * s + t;
    if (relu) x = fmaxf(x, 0.f);
    r[i] = x;
  }
  return r;
}

// mlp1 = relu(bn(W8x8 @ feat + b)) computed on the fly (feature fits in L2).
__device__ __forceinline__ void fpc8(const P &p, int b, int n, float *out8) {
  float f[8];
#pragma unroll
  for (int c = 0; c < 8; ++c)
    f[c] = p.feature[((long)b * 8 + c) * NPTS + n];
#pragma unroll
  for (int o = 0; o < 8; ++o) {
    float acc = p.mlp1_b[o];
#pragma unroll
    for (int c = 0; c < 8; ++c) acc += p.mlp1_w[o * 8 + c] * f[c];
    float g = p.mlp1_bn[o], be = p.mlp1_bn[8 + o], mu = p.mlp1_bn[16 + o],
          va = p.mlp1_bn[24 + o];
    float s = g * rsqrtf(va + 1e-5f);
    out8[o] = fmaxf(s * (acc - mu) + be, 0.f);
  }
}

__global__ __launch_bounds__(TPB) void fused_block(P p) {
  __shared__ _Float16 s_lw[512 * SL];         // lin_w^T staged f16 (~33 KB)
  __shared__ _Float16 s_tA[WAVES][32 * SA];   // per-wave staging tile
  __shared__ _Float16 s_tW[WAVES][16 * SW];   // per-wave WeightNet output
  __shared__ _Float16 s_tN[WAVES][8 * SN];    // per-wave [8 pts][512] "new"

  const int lane = threadIdx.x & 31;
  const int wv = threadIdx.x >> 5;
  const int col = lane & 15;
  const int m0 = (lane & 16) ? 8 : 0;

  __builtin_prefetch(p.lin_w, 0, 1);  // global_prefetch into L2
  __builtin_prefetch(p.c3_w, 0, 1);

  // Stage lin_w [32][512] -> LDS f16 [cw][o]
  for (int i = threadIdx.x; i < 512 * 32; i += TPB) {
    int o = i >> 9, cw = i & 511;
    s_lw[cw * SL + o] = (_Float16)p.lin_w[o * 512 + cw];
  }
  __syncthreads();

  _Float16 *tA = s_tA[wv];
  _Float16 *tW = s_tW[wv];
  _Float16 *tN = s_tN[wv];

  // Persistent B fragments (weights are tiny; L2-resident).
  v16h Bwn0 = make_B(p.wn0_w, 3, 8, lane, 0);
  v16h Bwn1 = make_B(p.wn1_w, 8, 8, lane, 0);
  v16h Bwn2 = make_B(p.wn2_w, 8, 16, lane, 0);
  v16h Bfc  = make_B(p.fc_w, 8, 8, lane, 0);
  v16h Bc0  = make_B(p.c0_w, 8, 16, lane, 0);
  v16h Bc1  = make_B(p.c1_w, 16, 16, lane, 0);
  v16h Bc2  = make_B(p.c2_w, 16, 16, lane, 0);
  v16h Bc3a = make_B(p.c3_w, 16, 32, lane, 0);
  v16h Bc3b = make_B(p.c3_w, 16, 32, lane, 16);
  v16h Bm2a = make_B(p.mlp2_w, 32, 32, lane, 0);
  v16h Bm2b = make_B(p.mlp2_w, 32, 32, lane, 16);
  v16h Bsca = make_B(p.sc_w, 8, 32, lane, 0);
  v16h Bscb = make_B(p.sc_w, 8, 32, lane, 16);

  // Per-lane BN affines (channel = C-layout column).
  float s_wn0, t_wn0, s_wn1, t_wn1, s_wn2, t_wn2;
  bn_aff(p.wn0_bn, 8, p.wn0_b, col, col < 8, s_wn0, t_wn0);
  bn_aff(p.wn1_bn, 8, p.wn1_b, col, col < 8, s_wn1, t_wn1);
  bn_aff(p.wn2_bn, 16, p.wn2_b, col, true, s_wn2, t_wn2);
  float s_c0, t_c0, s_c1, t_c1, s_c2, t_c2, s_c3a, t_c3a, s_c3b, t_c3b;
  bn_aff(p.c0_bn, 16, p.c0_b, col, true, s_c0, t_c0);
  bn_aff(p.c1_bn, 16, p.c1_b, col, true, s_c1, t_c1);
  bn_aff(p.c2_bn, 16, p.c2_b, col, true, s_c2, t_c2);
  bn_aff(p.c3_bn, 32, p.c3_b, col, true, s_c3a, t_c3a);
  bn_aff(p.c3_bn, 32, p.c3_b, col + 16, true, s_c3b, t_c3b);
  float s_l0, t_l0, s_l1, t_l1, s_m20, t_m20, s_m21, t_m21, s_s0, t_s0, s_s1,
      t_s1;
  bn_aff(p.lin_bn, 32, p.lin_b, col, true, s_l0, t_l0);
  bn_aff(p.lin_bn, 32, p.lin_b, col + 16, true, s_l1, t_l1);
  bn_aff(p.mlp2_bn, 32, p.mlp2_b, col, true, s_m20, t_m20);
  bn_aff(p.mlp2_bn, 32, p.mlp2_b, col + 16, true, s_m21, t_m21);
  bn_aff(p.sc_bn, 32, p.sc_b, col, true, s_s0, t_s0);
  bn_aff(p.sc_bn, 32, p.sc_b, col + 16, true, s_s1, t_s1);

  const long gw = (long)blockIdx.x * WAVES + wv;  // global wave id
  const long P0 = gw * 16;                        // first linear point index
  const int b = (int)(P0 / NPTS);
  const int n0 = (int)(P0 % NPTS);

  // Two 8-point groups per wave-tile.
  for (int g = 0; g < 2; ++g) {
    const int nG = n0 + g * 8;

    // ---- per-point stages: M = 16 neighbors ----
    for (int pp = 0; pp < 8; ++pp) {
      const int n = nG + pp;
      int nb = 0;
      if (lane < 16) nb = p.nidx[((long)b * NPTS + n) * KNB + lane];

      // relative xyz -> tA[k][0..2]
      if (lane < 16) {
        const float *xb = p.xyz + (long)b * NPTS * 3;
#pragma unroll
        for (int c = 0; c < 3; ++c)
          tA[lane * SA + c] = (_Float16)(xb[n * 3 + c] - xb[nb * 3 + c]);
      }

      // WeightNet: 3 -> 8 -> 8 -> 16
      v16h a = load_A(tA, SA, lane, 3);
      v8f c = wmma(a, Bwn0, zeroC());
      c = aff(c, s_wn0, t_wn0, true);
      store_C(tA, SA, lane, c);
      a = load_A(tA, SA, lane, 8);
      c = wmma(a, Bwn1, zeroC());
      c = aff(c, s_wn1, t_wn1, true);
      store_C(tA, SA, lane, c);
      a = load_A(tA, SA, lane, 8);
      c = wmma(a, Bwn2, zeroC());
      c = aff(c, s_wn2, t_wn2, true);
      store_C(tW, SW, lane, c);  // w_mat [k][16]

      // relative features -> tA[k][0..7]
      if (lane < 16) {
        float fcen[8], fngh[8];
        fpc8(p, b, n, fcen);
        fpc8(p, b, nb, fngh);
#pragma unroll
        for (int ch = 0; ch < 8; ++ch)
          tA[lane * SA + ch] = (_Float16)(fcen[ch] - fngh[ch]);
      }

      // attention: scores = softmax_k(fc(rel_f)); fr = scores * rel_f
      a = load_A(tA, SA, lane, 8);
      v8f sv = wmma(a, Bfc, zeroC());
      float mx = -1e30f;
#pragma unroll
      for (int r = 0; r < 8; ++r) mx = fmaxf(mx, sv[r]);
      mx = fmaxf(mx, __shfl_xor(mx, 16, 32));
      float sm = 0.f;
      v8f ev;
#pragma unroll
      for (int r = 0; r < 8; ++r) { ev[r] = expf(sv[r] - mx); sm += ev[r]; }
      sm += __shfl_xor(sm, 16, 32);
      float inv = 1.f / sm;
      float frv[8];
#pragma unroll
      for (int r = 0; r < 8; ++r)
        frv[r] = (float)tA[(m0 + r) * SA + col] * (ev[r] * inv);
#pragma unroll
      for (int r = 0; r < 8; ++r) tA[(m0 + r) * SA + col] = (_Float16)frv[r];

      // mlp_convs: 8 -> 16 -> 16 -> 16 -> 32
      a = load_A(tA, SA, lane, 8);
      c = wmma(a, Bc0, zeroC());
      c = aff(c, s_c0, t_c0, true);
      store_C(tA, SA, lane, c);
      a = load_A(tA, SA, lane, 16);
      c = wmma(a, Bc1, zeroC());
      c = aff(c, s_c1, t_c1, true);
      store_C(tA, SA, lane, c);
      a = load_A(tA, SA, lane, 16);
      c = wmma(a, Bc2, zeroC());
      c = aff(c, s_c2, t_c2, true);
      store_C(tA, SA, lane, c);
      a = load_A(tA, SA, lane, 16);
      v8f ca = wmma(a, Bc3a, zeroC());
      ca = aff(ca, s_c3a, t_c3a, true);
      v8f cb = wmma(a, Bc3b, zeroC());
      cb = aff(cb, s_c3b, t_c3b, true);

      // transpose fr into tA[ch][k] (ch = 0..31)
#pragma unroll
      for (int r = 0; r < 8; ++r) tA[col * SA + (m0 + r)] = (_Float16)ca[r];
#pragma unroll
      for (int r = 0; r < 8; ++r)
        tA[(16 + col) * SA + (m0 + r)] = (_Float16)cb[r];

      // reduce over K: new[c][w] = sum_k fr[c][k] * w_mat[k][w]
      v16h bw = load_Bt(tW, SW, lane, 16);
      v16h a0 = load_A(tA, SA, lane, 16);
      v16h a1 = load_A(tA + 16 * SA, SA, lane, 16);
      v8f r0 = wmma(a0, bw, zeroC());
      v8f r1 = wmma(a1, bw, zeroC());
#pragma unroll
      for (int r = 0; r < 8; ++r)
        tN[pp * SN + (m0 + r) * 16 + col] = (_Float16)r0[r];
#pragma unroll
      for (int r = 0; r < 8; ++r)
        tN[pp * SN + (16 + m0 + r) * 16 + col] = (_Float16)r1[r];
    }

    // ---- 8-point batched Linear 512 -> 32 (16x512x32 GEMM, rows 8..15 pad) ----
    const int m = lane & 15;
    v8f l0 = zeroC(), l1 = zeroC();
#pragma unroll
    for (int kk = 0; kk < 16; ++kk) {
      v16h aL;
#pragma unroll
      for (int h = 0; h < 16; ++h)
        aL[h] = (m < 8) ? tN[m * SN + kk * 32 + a_k_of(lane, h)]
                        : (_Float16)0.0f;
      v16h b0, b1;
#pragma unroll
      for (int h = 0; h < 16; ++h) {
        int k = ((lane & 16) ? 16 : 0) + h;
        b0[h] = s_lw[(kk * 32 + k) * SL + col];
        b1[h] = s_lw[(kk * 32 + k) * SL + 16 + col];
      }
      l0 = wmma(aL, b0, l0);
      l1 = wmma(aL, b1, l1);
    }
    l0 = aff(l0, s_l0, t_l0, false);  // bias + BN1d (no relu)
    l1 = aff(l1, s_l1, t_l1, false);

    // stage "new" [pt][32] for mlp2
#pragma unroll
    for (int r = 0; r < 8; ++r) tA[(m0 + r) * SA + col] = (_Float16)l0[r];
#pragma unroll
    for (int r = 0; r < 8; ++r) tA[(m0 + r) * SA + 16 + col] = (_Float16)l1[r];

    // mlp2 32 -> 32 (no activation)
    v16h a2 = load_A(tA, SA, lane, 32);
    v8f f20 = aff(wmma(a2, Bm2a, zeroC()), s_m20, t_m20, false);
    v8f f21 = aff(wmma(a2, Bm2b, zeroC()), s_m21, t_m21, false);

    // shortcut 8 -> 32 on raw features (rows 0..7 = this group's points)
    if (lane < 8) {
      int nn = nG + lane;
#pragma unroll
      for (int ch = 0; ch < 8; ++ch)
        tA[lane * SA + ch] =
            (_Float16)p.feature[((long)b * 8 + ch) * NPTS + nn];
    }
    v16h asc = load_A(tA, SA, lane, 8);
    v8f sc0 = aff(wmma(asc, Bsca, zeroC()), s_s0, t_s0, false);
    v8f sc1 = aff(wmma(asc, Bscb, zeroC()), s_s1, t_s1, false);

    // leaky_relu(f2 + sc, 0.2) -> out[b][o][n]; rows 0..7 valid
    float *ob = p.out + (long)b * 32 * NPTS;
#pragma unroll
    for (int r = 0; r < 8; ++r) {
      int ptv = m0 + r;
      if (ptv < 8) {
        float v0 = f20[r] + sc0[r];
        v0 = v0 > 0.f ? v0 : 0.2f * v0;
        float v1 = f21[r] + sc1[r];
        v1 = v1 > 0.f ? v1 : 0.2f * v1;
        ob[(long)col * NPTS + nG + ptv] = v0;
        ob[(long)(col + 16) * NPTS + nG + ptv] = v1;
      }
    }
  }
}

extern "C" void kernel_launch(void *const *d_in, const int *in_sizes, int n_in,
                              void *d_out, int out_size, void *d_ws,
                              size_t ws_size, hipStream_t stream) {
  (void)in_sizes; (void)n_in; (void)d_ws; (void)ws_size; (void)out_size;
  P prm;
  prm.feature = (const float *)d_in[0];
  prm.xyz     = (const float *)d_in[1];
  prm.mlp1_w  = (const float *)d_in[2];
  prm.mlp1_b  = (const float *)d_in[3];
  prm.mlp1_bn = (const float *)d_in[4];
  prm.wn0_w   = (const float *)d_in[5];
  prm.wn0_b   = (const float *)d_in[6];
  prm.wn0_bn  = (const float *)d_in[7];
  prm.wn1_w   = (const float *)d_in[8];
  prm.wn1_b   = (const float *)d_in[9];
  prm.wn1_bn  = (const float *)d_in[10];
  prm.wn2_w   = (const float *)d_in[11];
  prm.wn2_b   = (const float *)d_in[12];
  prm.wn2_bn  = (const float *)d_in[13];
  prm.fc_w    = (const float *)d_in[14];
  prm.c0_w    = (const float *)d_in[15];
  prm.c0_b    = (const float *)d_in[16];
  prm.c0_bn   = (const float *)d_in[17];
  prm.c1_w    = (const float *)d_in[18];
  prm.c1_b    = (const float *)d_in[19];
  prm.c1_bn   = (const float *)d_in[20];
  prm.c2_w    = (const float *)d_in[21];
  prm.c2_b    = (const float *)d_in[22];
  prm.c2_bn   = (const float *)d_in[23];
  prm.c3_w    = (const float *)d_in[24];
  prm.c3_b    = (const float *)d_in[25];
  prm.c3_bn   = (const float *)d_in[26];
  prm.lin_w   = (const float *)d_in[27];
  prm.lin_b   = (const float *)d_in[28];
  prm.lin_bn  = (const float *)d_in[29];
  prm.mlp2_w  = (const float *)d_in[30];
  prm.mlp2_b  = (const float *)d_in[31];
  prm.mlp2_bn = (const float *)d_in[32];
  prm.sc_w    = (const float *)d_in[33];
  prm.sc_b    = (const float *)d_in[34];
  prm.sc_bn   = (const float *)d_in[35];
  prm.nidx    = (const int *)d_in[36];
  prm.out     = (float *)d_out;

  const int totalWaveTiles = (BATCH * NPTS) / 16;  // 8192
  const int blocks = totalWaveTiles / WAVES;       // 2048
  fused_block<<<blocks, TPB, 0, stream>>>(prm);
}
